// HelloBitNetChoc_76879914598603
// MI455X (gfx1250) — matrix operand order repaired and checked by reading, not compiled, not run on hardware
//
#include <hip/hip_runtime.h>
#include <hip/hip_bf16.h>
#include <cstdint>

#define SEQ  4096
#define DIM  2048
#define HID  8192

typedef __attribute__((ext_vector_type(16))) __bf16 v16bf;
typedef __attribute__((ext_vector_type(8)))  float  v8f;

// ---------- async-to-LDS availability (compile-safe probe) ----------
#if defined(__has_builtin)
#  if __has_builtin(__builtin_amdgcn_global_load_async_to_lds_b128)
#    define HAS_ASYNC 1
#  else
#    define HAS_ASYNC 0
#  endif
#else
#  define HAS_ASYNC 0
#endif

#if HAS_ASYNC
typedef int v4i __attribute__((__vector_size__(4 * sizeof(int))));
typedef __attribute__((address_space(1))) v4i* as1_v4i;
typedef __attribute__((address_space(3))) v4i* as3_v4i;
#  if __has_builtin(__builtin_amdgcn_s_wait_asynccnt)
#    define WAIT_ASYNC() __builtin_amdgcn_s_wait_asynccnt(0)
#  else
#    define WAIT_ASYNC() asm volatile("s_wait_asynccnt 0x0" ::: "memory")
#  endif
#else
#  define WAIT_ASYNC()
#endif

// ---------- scalar helpers ----------
__device__ __forceinline__ unsigned short f32_to_bf16_rn(float f) {
  unsigned int u = __float_as_uint(f);
  unsigned int r = 0x7FFFu + ((u >> 16) & 1u);
  return (unsigned short)((u + r) >> 16);
}

// ---------- elementwise prep kernels ----------
__global__ void k_sign_bf16(const float* __restrict__ w, unsigned short* __restrict__ o, int n) {
  int i = blockIdx.x * blockDim.x + threadIdx.x;
  if (i < n) {
    float x = w[i];
    float s = (x > 0.f) ? 1.f : ((x < 0.f) ? -1.f : 0.f);
    o[i] = f32_to_bf16_rn(s);
  }
}

__global__ void k_f32_bf16(const float* __restrict__ a, unsigned short* __restrict__ o, int n) {
  int i = blockIdx.x * blockDim.x + threadIdx.x;
  if (i < n) o[i] = f32_to_bf16_rn(a[i]);
}

// LDS-tiled transpose: in[R][C] f32 -> out[C][R] bf16 (optionally sign-quantized).
// 64x64 tiles; f32 reads and bf16 writes both 128-bit coalesced.
template<int SIGN>
__global__ void k_trans_bf16(const float* __restrict__ in, unsigned short* __restrict__ out,
                             int R, int C) {
  __shared__ unsigned short t[64][72];
  const int r0 = blockIdx.y * 64, c0 = blockIdx.x * 64;
  const int tid = threadIdx.x;
  #pragma unroll
  for (int v = tid; v < 64 * 16; v += 256) {       // 16 float4 per row
    int r = v >> 4, j = v & 15;
    float4 f = ((const float4*)(in + (size_t)(r0 + r) * C + c0))[j];
    float x[4] = {f.x, f.y, f.z, f.w};
    #pragma unroll
    for (int e = 0; e < 4; ++e) {
      float y = SIGN ? ((x[e] > 0.f) ? 1.f : ((x[e] < 0.f) ? -1.f : 0.f)) : x[e];
      t[r][j * 4 + e] = f32_to_bf16_rn(y);
    }
  }
  __syncthreads();
  #pragma unroll
  for (int v = tid; v < 64 * 8; v += 256) {        // 8 uint4 per output row
    int c = v >> 3, j = v & 7;
    union { unsigned short u[8]; uint4 q; } pk;
    #pragma unroll
    for (int e = 0; e < 8; ++e) pk.u[e] = t[j * 8 + e][c];
    ((uint4*)(out + (size_t)(c0 + c) * R + r0))[j] = pk.q;
  }
}

// ---------- row softmax: f32 scores -> bf16 probs ----------
__global__ void k_softmax(const float* __restrict__ S, unsigned short* __restrict__ P, int ncol) {
  __shared__ float red[256];
  const int row = blockIdx.x;
  const int tid = threadIdx.x;
  const float* srow = S + (size_t)row * ncol;

  float m = -3.4e38f;
  for (int i = tid; i < ncol; i += 256) m = fmaxf(m, srow[i]);
  red[tid] = m; __syncthreads();
  for (int s = 128; s > 0; s >>= 1) {
    if (tid < s) red[tid] = fmaxf(red[tid], red[tid + s]);
    __syncthreads();
  }
  m = red[0]; __syncthreads();

  float sum = 0.f;
  for (int i = tid; i < ncol; i += 256) sum += __expf(srow[i] - m);
  red[tid] = sum; __syncthreads();
  for (int s = 128; s > 0; s >>= 1) {
    if (tid < s) red[tid] += red[tid + s];
    __syncthreads();
  }
  float inv = 1.f / red[0];

  unsigned short* prow = P + (size_t)row * ncol;
  for (int i = tid; i < ncol; i += 256) prow[i] = f32_to_bf16_rn(__expf(srow[i] - m) * inv);
}

// ---------- WMMA bf16 GEMM (canonical layouts) ----------
// C[M,N] = A @ B^T with A[M,K] row-major, B[N,K] row-major, f32 accumulate.
// Block: 256 threads = 8 waves; block tile 256x128; wave tile 64x64 (16 WMMA / K-step).
// Double-buffered LDS staging, async global->LDS when available.
// EPI: 0 scale->f32 | 1 bf16 row+T | 2 relu bf16 row+T
//      3 pred: f32 C + bf16(C-auxF) row+T | 4 mask(auxB>0) -> T only | 5 sgd: outF=auxF-lr*C
union FragBF {
  struct { uint4 lo, hi; } q;
  v16bf v;
};

template<int EPI>
__launch_bounds__(256)
__global__ void k_gemm(const unsigned short* __restrict__ A,
                       const unsigned short* __restrict__ B,
                       int M, int N, int K,
                       float* __restrict__ outF, unsigned short* __restrict__ outB,
                       unsigned short* __restrict__ outT,
                       const float* __restrict__ auxF, const unsigned short* __restrict__ auxB,
                       const float* __restrict__ lrp, float scale) {
  constexpr int BM = 256, BN = 128, KT = 32;
  constexpr int LS = KT + 8;                       // padded LDS row stride (ushorts)
  __shared__ unsigned short As[2][BM * LS];        // As[buf][m][k]
  __shared__ unsigned short Bs[2][BN * LS];        // Bs[buf][n][k]

  const int tid  = threadIdx.x;
  const int wave = tid >> 5;
  const int lane = tid & 31;
  const int half = lane >> 4;     // 0/1
  const int lrow = lane & 15;
  const int mr   = (wave >> 1) * 64;   // wave tile origin in block
  const int nr   = (wave & 1) * 64;
  const int blockN0 = blockIdx.x * BN;
  const int blockM0 = blockIdx.y * BM;

  v8f acc[16];
  #pragma unroll
  for (int i = 0; i < 16; ++i) acc[i] = (v8f){};

  auto fill = [&](int buf, int kt) {
    const int k0 = kt * KT;
    #pragma unroll
    for (int v = tid; v < BM * 4; v += 256) {      // 4 uint4 per A row
      int m = v >> 2, j = v & 3;
      const uint4* g = (const uint4*)(A + (size_t)(blockM0 + m) * K + k0) + j;
      uint4* l = (uint4*)(&As[buf][m * LS]) + j;
#if HAS_ASYNC
      __builtin_amdgcn_global_load_async_to_lds_b128((as1_v4i)g, (as3_v4i)l, 0, 0);
#else
      *l = *g;
#endif
    }
    #pragma unroll
    for (int v = tid; v < BN * 4; v += 256) {      // 4 uint4 per B row
      int n = v >> 2, j = v & 3;
      const uint4* g = (const uint4*)(B + (size_t)(blockN0 + n) * K + k0) + j;
      uint4* l = (uint4*)(&Bs[buf][n * LS]) + j;
#if HAS_ASYNC
      __builtin_amdgcn_global_load_async_to_lds_b128((as1_v4i)g, (as3_v4i)l, 0, 0);
#else
      *l = *g;
#endif
    }
  };

  const int ksteps = K / KT;
  fill(0, 0);
  WAIT_ASYNC();
  __syncthreads();

  for (int kt = 0; kt < ksteps; ++kt) {
    const int buf = kt & 1;
    if (kt + 1 < ksteps) fill(buf ^ 1, kt + 1);    // overlap next-tile staging with WMMA

    FragBF a[4], b[4];
    #pragma unroll
    for (int i = 0; i < 4; ++i) {
      // A frag (ISA 7.12.2): lane holds row mr+16i+lrow, K chunks [8h,8h+8) and [16+8h,..)
      const uint4* p = (const uint4*)(&As[buf][(mr + i * 16 + lrow) * LS]);
      a[i].q.lo = p[half];     a[i].q.hi = p[2 + half];
      // B frag: lane holds col nr+16i+lrow, K chunks [8h,8h+8) and [16+8h,..)
      const uint4* q = (const uint4*)(&Bs[buf][(nr + i * 16 + lrow) * LS]);
      b[i].q.lo = q[half]; b[i].q.hi = q[2 + half];
    }
    #pragma unroll
    for (int i = 0; i < 4; ++i)
      #pragma unroll
      for (int j = 0; j < 4; ++j)
        acc[i * 4 + j] = __builtin_amdgcn_wmma_f32_16x16x32_bf16(
            false, a[i].v, false, b[j].v, (short)0, acc[i * 4 + j], false, false);

    WAIT_ASYNC();
    __syncthreads();
  }

  // ---- epilogue ----
  // D layout: lane holds col gn, 8 consecutive rows gm0..gm0+7 -> transposed bf16
  // output is one coalesced b128 store per 16x16 tile per lane.
  const float lr = (EPI == 5) ? lrp[0] : 0.f;
  #pragma unroll
  for (int ti = 0; ti < 4; ++ti) {
    #pragma unroll
    for (int tj = 0; tj < 4; ++tj) {
      v8f c = acc[ti * 4 + tj];
      const int gm0 = blockM0 + mr + ti * 16 + half * 8;
      const int gn  = blockN0 + nr + tj * 16 + lrow;
      union { unsigned short u[8]; uint4 q; } pk;
      #pragma unroll
      for (int r = 0; r < 8; ++r) {
        size_t idx = (size_t)(gm0 + r) * (size_t)N + (size_t)gn;
        float v = c[r];
        if (EPI == 0) {
          outF[idx] = v * scale;
        } else if (EPI == 1) {
          unsigned short h = f32_to_bf16_rn(v);
          outB[idx] = h; pk.u[r] = h;
        } else if (EPI == 2) {
          unsigned short h = f32_to_bf16_rn(fmaxf(v, 0.f));
          outB[idx] = h; pk.u[r] = h;
        } else if (EPI == 3) {
          outF[idx] = v;
          unsigned short h = f32_to_bf16_rn(v - auxF[idx]);
          outB[idx] = h; pk.u[r] = h;
        } else if (EPI == 4) {
          unsigned short mk = auxB[idx];
          pk.u[r] = (mk != 0 && (mk & 0x8000u) == 0) ? f32_to_bf16_rn(v) : (unsigned short)0;
        } else {
          outF[idx] = auxF[idx] - lr * v;
        }
      }
      if (EPI == 1 || EPI == 2 || EPI == 3 || EPI == 4) {
        *(uint4*)(outT + (size_t)gn * (size_t)M + (size_t)gm0) = pk.q;
      }
    }
  }
}

// ---------- launch ----------
extern "C" void kernel_launch(void* const* d_in, const int* in_sizes, int n_in,
                              void* d_out, int out_size, void* d_ws, size_t ws_size,
                              hipStream_t stream) {
  (void)in_sizes; (void)n_in; (void)out_size; (void)ws_size;

  const float* src    = (const float*)d_in[0];
  const float* tgt    = (const float*)d_in[1];
  const float* w_qkv  = (const float*)d_in[2];
  const float* w_ffn1 = (const float*)d_in[3];
  const float* w_ffn2 = (const float*)d_in[4];
  const float* lr     = (const float*)d_in[5];

  float* out_pred  = (float*)d_out;
  float* out_wqkv  = out_pred  + (size_t)SEQ * DIM;
  float* out_wffn1 = out_wqkv  + (size_t)DIM * DIM;
  float* out_wffn2 = out_wffn1 + (size_t)HID * DIM;

  // workspace carve
  char* ws = (char*)d_ws;
  auto carve = [&](size_t bytes) { char* p = ws; ws += bytes; return p; };
  unsigned short* src_b   = (unsigned short*)carve((size_t)SEQ * DIM * 2);
  unsigned short* src_T   = (unsigned short*)carve((size_t)DIM * SEQ * 2);
  unsigned short* wqkv_b  = (unsigned short*)carve((size_t)DIM * DIM * 2);
  unsigned short* wffn1_b = (unsigned short*)carve((size_t)HID * DIM * 2);
  unsigned short* wffn2_b = (unsigned short*)carve((size_t)DIM * HID * 2);
  unsigned short* wffn2_T = (unsigned short*)carve((size_t)HID * DIM * 2);
  unsigned short* qkv_b   = (unsigned short*)carve((size_t)SEQ * DIM * 2);
  unsigned short* qkv_T   = (unsigned short*)carve((size_t)DIM * SEQ * 2);
  unsigned short* probs_b = (unsigned short*)carve((size_t)SEQ * SEQ * 2);
  unsigned short* ctx_b   = (unsigned short*)carve((size_t)SEQ * DIM * 2);
  unsigned short* ctx_T   = (unsigned short*)carve((size_t)DIM * SEQ * 2);
  unsigned short* ffnh_b  = (unsigned short*)carve((size_t)SEQ * HID * 2);
  unsigned short* ffnh_T  = (unsigned short*)carve((size_t)HID * SEQ * 2);
  unsigned short* lg_b    = (unsigned short*)carve((size_t)SEQ * DIM * 2);
  unsigned short* lg_T    = (unsigned short*)carve((size_t)DIM * SEQ * 2);
  float*          scores  = (float*)carve((size_t)SEQ * SEQ * 4);
  unsigned short* gfh_T   = (unsigned short*)scores;  // alias: scores dead after softmax

  dim3 blk(256);
  auto g1 = [](size_t n) { return dim3((unsigned)((n + 255) / 256)); };

  // prep: bf16 conversions, sign quantization, and the two source-side transposes
  k_f32_bf16 <<<g1((size_t)SEQ * DIM), blk, 0, stream>>>(src,    src_b,   SEQ * DIM);
  k_sign_bf16<<<g1((size_t)DIM * DIM), blk, 0, stream>>>(w_qkv,  wqkv_b,  DIM * DIM);
  k_sign_bf16<<<g1((size_t)HID * DIM), blk, 0, stream>>>(w_ffn1, wffn1_b, HID * DIM);
  k_sign_bf16<<<g1((size_t)DIM * HID), blk, 0, stream>>>(w_ffn2, wffn2_b, DIM * HID);
  k_trans_bf16<0><<<dim3(DIM / 64, SEQ / 64), blk, 0, stream>>>(src,    src_T,   SEQ, DIM);
  k_trans_bf16<1><<<dim3(HID / 64, DIM / 64), blk, 0, stream>>>(w_ffn2, wffn2_T, DIM, HID);

  // 1) qkv = src @ Wq^T                       [SEQ,DIM] (+ qkv^T)
  k_gemm<1><<<dim3(DIM / 128, SEQ / 256), blk, 0, stream>>>(
      src_b, wqkv_b, SEQ, DIM, DIM,
      nullptr, qkv_b, qkv_T, nullptr, nullptr, nullptr, 0.f);

  // 2) scores = (qkv @ qkv^T) / 11.31         [SEQ,SEQ] f32
  k_gemm<0><<<dim3(SEQ / 128, SEQ / 256), blk, 0, stream>>>(
      qkv_b, qkv_b, SEQ, SEQ, DIM,
      scores, nullptr, nullptr, nullptr, nullptr, nullptr, 1.0f / 11.31f);

  // 3) softmax rows -> bf16 probs
  k_softmax<<<dim3(SEQ), blk, 0, stream>>>(scores, probs_b, SEQ);

  // 4) context = probs @ qkv = probs @ (qkv^T)^T   [SEQ,DIM] (+ ctx^T)
  k_gemm<1><<<dim3(DIM / 128, SEQ / 256), blk, 0, stream>>>(
      probs_b, qkv_T, SEQ, DIM, SEQ,
      nullptr, ctx_b, ctx_T, nullptr, nullptr, nullptr, 0.f);

  // 5) ffn_h = relu(context @ W1^T)           [SEQ,HID] (+ ffn_h^T)
  k_gemm<2><<<dim3(HID / 128, SEQ / 256), blk, 0, stream>>>(
      ctx_b, wffn1_b, SEQ, HID, DIM,
      nullptr, ffnh_b, ffnh_T, nullptr, nullptr, nullptr, 0.f);

  // 6) pred = ffn_h @ W2^T ; loss_grad = pred - tgt  [SEQ,DIM] (+ lg^T)
  k_gemm<3><<<dim3(DIM / 128, SEQ / 256), blk, 0, stream>>>(
      ffnh_b, wffn2_b, SEQ, DIM, HID,
      out_pred, lg_b, lg_T, tgt, nullptr, nullptr, 0.f);

  // 7) grad_ffn_h = (ffn_h>0) ? loss_grad @ W2 : 0  -> stored transposed only [HID,SEQ]
  k_gemm<4><<<dim3(HID / 128, SEQ / 256), blk, 0, stream>>>(
      lg_b, wffn2_T, SEQ, HID, DIM,
      nullptr, nullptr, gfh_T, nullptr, ffnh_b, nullptr, 0.f);

  // 8) new_w_ffn2 = w_ffn2 - lr * (lg^T @ ffn_h)    [DIM,HID]
  k_gemm<5><<<dim3(HID / 128, DIM / 256), blk, 0, stream>>>(
      lg_T, ffnh_T, DIM, HID, SEQ,
      out_wffn2, nullptr, nullptr, w_ffn2, nullptr, lr, 0.f);

  // 9) new_w_ffn1 = w_ffn1 - lr * (gfh^T @ ctx)     [HID,DIM]
  k_gemm<5><<<dim3(DIM / 128, HID / 256), blk, 0, stream>>>(
      gfh_T, ctx_T, HID, DIM, SEQ,
      out_wffn1, nullptr, nullptr, w_ffn1, nullptr, lr, 0.f);

  // 10) new_w_qkv = w_qkv - lr * (lg^T @ src)       [DIM,DIM]
  k_gemm<5><<<dim3(DIM / 128, DIM / 256), blk, 0, stream>>>(
      lg_T, src_T, DIM, DIM, SEQ,
      out_wqkv, nullptr, nullptr, w_qkv, nullptr, lr, 0.f);
}
